// SpatialCrossAttention_28071906247021
// MI455X (gfx1250) — matrix operand order repaired and checked by reading, not compiled, not run on hardware
//
#include <hip/hip_runtime.h>
#include <hip/hip_bf16.h>
#include <math.h>

typedef _Float16 v16h __attribute__((ext_vector_type(16)));
typedef float    v8f  __attribute__((ext_vector_type(8)));

#define CAMS   6
#define NQ     4096
#define EMBED  256
#define HEADS  8
#define LTOT   19560   // 92*160 + 46*80 + 23*40 + 12*20

// ---------------------------------------------------------------------------
// Prep: build fp16 [N][K] transposed copies of the weight matrices so that a
// WMMA B-fragment (16 halves, K = h*16..h*16+15 at fixed column n) is one
// contiguous 32B load. Also fuse b_off|b_attn into one 768-entry bias.
// ---------------------------------------------------------------------------
__global__ __launch_bounds__(256) void prep_kernel(
    const float* __restrict__ Wv,  const float* __restrict__ Woff,
    const float* __restrict__ Wat, const float* __restrict__ boff,
    const float* __restrict__ bat, const float* __restrict__ Wo,
    _Float16* __restrict__ Wv16, _Float16* __restrict__ Woa16,
    _Float16* __restrict__ Wo16, float* __restrict__ bias_oa)
{
    int i = blockIdx.x * 256 + threadIdx.x;
    if (i < 65536) {                               // W_value: 256x256 -> [n][k]
        int n = i >> 8, k = i & 255;
        Wv16[i] = (_Float16)Wv[k * 256 + n];
    } else if (i < 65536 + 196608) {               // [W_off | W_attn]: 768x256
        int j = i - 65536;
        int n = j >> 8, k = j & 255;
        Woa16[j] = (_Float16)((n < 512) ? Woff[k * 512 + n]
                                        : Wat[k * 256 + (n - 512)]);
    } else if (i < 65536 + 196608 + 65536) {       // W_out: 256x256 -> [n][k]
        int j = i - 65536 - 196608;
        int n = j >> 8, k = j & 255;
        Wo16[j] = (_Float16)Wo[k * 256 + n];
    } else if (i < 65536 + 196608 + 65536 + 768) { // fused bias
        int n = i - 65536 - 196608 - 65536;
        bias_oa[n] = (n < 512) ? boff[n] : bat[n - 512];
    }
}

// ---------------------------------------------------------------------------
// WMMA GEMM, K fixed = 256.  C[M,N] = A[M,256] * B16[N,256]^T + bias (+res).
// Block = 256 threads = 8 waves; each wave computes a 16x64 strip of C
// (4x v_wmma_f32_16x16x32_f16 accumulators, 8 K-steps).
// A-fragment loaded directly from fp32 global and converted (layout:
// lanes 0-15 = rows M, half h=lane>>4 selects K in {h*16..h*16+15}).
// B-fragment: 16 contiguous halves from the [N][K] fp16 copy.
// ---------------------------------------------------------------------------
__global__ __launch_bounds__(256) void wmma_gemm_k256(
    const float* __restrict__ A, const _Float16* __restrict__ B16,
    const float* __restrict__ bias, const float* __restrict__ residual,
    float* __restrict__ C32, _Float16* __restrict__ C16,
    int M, int N, size_t strideAz, size_t strideCz)
{
    A += (size_t)blockIdx.z * strideAz;
    const int wave  = threadIdx.x >> 5;
    const int lane  = threadIdx.x & 31;
    const int hh    = lane >> 4;      // half-wave id
    const int ml    = lane & 15;      // A row within tile / B column within tile
    const int mbase = blockIdx.x * 128 + wave * 16;
    const int nbase = blockIdx.y * 64;
    const int row   = mbase + ml;
    const bool rok  = row < M;
    const float* arow = A + (size_t)row * 256;

    v8f acc[4] = {};
    for (int kb = 0; kb < 256; kb += 32) {
        float tmp[16];
        if (rok) {
            const int k0 = kb + hh * 16;
            float4 p0 = *(const float4*)(arow + k0);
            float4 p1 = *(const float4*)(arow + k0 + 4);
            float4 p2 = *(const float4*)(arow + k0 + 8);
            float4 p3 = *(const float4*)(arow + k0 + 12);
            tmp[0]=p0.x; tmp[1]=p0.y; tmp[2]=p0.z; tmp[3]=p0.w;
            tmp[4]=p1.x; tmp[5]=p1.y; tmp[6]=p1.z; tmp[7]=p1.w;
            tmp[8]=p2.x; tmp[9]=p2.y; tmp[10]=p2.z; tmp[11]=p2.w;
            tmp[12]=p3.x; tmp[13]=p3.y; tmp[14]=p3.z; tmp[15]=p3.w;
        } else {
            #pragma unroll
            for (int i = 0; i < 16; ++i) tmp[i] = 0.f;
        }
        v16h a;
        #pragma unroll
        for (int i = 0; i < 16; ++i) a[i] = (_Float16)tmp[i];

        #pragma unroll
        for (int t = 0; t < 4; ++t) {
            const _Float16* bp =
                B16 + (size_t)(nbase + t * 16 + ml) * 256 + kb + hh * 16;
            v16h b = *(const v16h*)bp;
            acc[t] = __builtin_amdgcn_wmma_f32_16x16x32_f16(
                false, a, false, b, (short)0, acc[t], false, false);
        }
    }

    _Float16* o16 = C16 ? C16 + (size_t)blockIdx.z * strideCz : (_Float16*)0;
    float*    o32 = C32 ? C32 + (size_t)blockIdx.z * strideCz : (float*)0;
    #pragma unroll
    for (int t = 0; t < 4; ++t) {
        const int n = nbase + t * 16 + ml;
        const float bi = bias ? bias[n] : 0.f;
        #pragma unroll
        for (int r = 0; r < 8; ++r) {
            // C layout: lanes 0-15 -> M = r, lanes 16-31 -> M = 8 + r
            const int m = mbase + hh * 8 + r;
            if (m < M) {
                float v = acc[t][r] + bi;
                size_t idx = (size_t)m * N + n;
                if (residual) v += residual[idx];
                if (o16) o16[idx] = (_Float16)v;
                else     o32[idx] = v;
            }
        }
    }
}

// ---------------------------------------------------------------------------
// Softmax over LEVELS*POINTS=32 per (q, head). One wave per (q,h); lane = lp.
// ---------------------------------------------------------------------------
__global__ __launch_bounds__(256) void softmax_kernel(
    const float* __restrict__ offattn, float* __restrict__ attn)
{
    const int q = blockIdx.x;
    const int h = threadIdx.x >> 5;
    const int lane = threadIdx.x & 31;
    float v = offattn[(size_t)q * 768 + 512 + h * 32 + lane];
    float m = v;
    #pragma unroll
    for (int o = 16; o > 0; o >>= 1) m = fmaxf(m, __shfl_xor(m, o, 32));
    float e = __expf(v - m);
    float s = e;
    #pragma unroll
    for (int o = 16; o > 0; o >>= 1) s += __shfl_xor(s, o, 32);
    attn[((size_t)q * HEADS + h) * 32 + lane] = e / s;
}

// ---------------------------------------------------------------------------
// Deformable bilinear gather. One wave per (q, head), lane = head-dim channel.
// val is fp16 [cam][L][head][32] so each corner reads a 64B contiguous chunk
// (mostly L2-resident: 60MB vs 192MB L2). Scalar params are wave-uniform.
// ---------------------------------------------------------------------------
__global__ __launch_bounds__(256) void sample_kernel(
    const _Float16* __restrict__ val,   // [CAMS][LTOT][256] fp16
    const float* __restrict__ offr,     // [NQ][768] (bias included)
    const float* __restrict__ attn,     // [NQ][HEADS][32]
    const float* __restrict__ ref,      // [CAMS][NQ][4][2]
    float* __restrict__ out_cam)        // [CAMS][NQ][256]
{
    constexpr int LVL_H[4] = {92, 46, 23, 12};
    constexpr int LVL_W[4] = {160, 80, 40, 20};
    constexpr int LSI[4]   = {0, 14720, 18400, 19320};

    const int q    = blockIdx.x;
    const int cam  = blockIdx.y;
    const int h    = threadIdx.x >> 5;
    const int lane = threadIdx.x & 31;

    const _Float16* vb = val + (size_t)cam * LTOT * 256 + h * 32 + lane;
    const float* oq = offr + (size_t)q * 768;
    const float* aq = attn + ((size_t)q * HEADS + h) * 32;
    const float* rq = ref  + ((size_t)cam * NQ + q) * 8;

    float acc = 0.f;
    #pragma unroll
    for (int lvl = 0; lvl < 4; ++lvl) {
        const int   Wi = LVL_W[lvl], Hi = LVL_H[lvl];
        const float Wf = (float)Wi,  Hf = (float)Hi;
        const int   base = LSI[lvl];
        for (int p = 0; p < 8; ++p) {
            const int col = ((h * 4 + lvl) * 8 + p) * 2;
            const float ox = oq[col], oy = oq[col + 1];
            const float aw = aq[lvl * 8 + p];
            const int dref = p & 3;                     // POINTS -> (ppz, D)
            const float x = (rq[dref * 2 + 0] + ox / Wf) * Wf - 0.5f;
            const float y = (rq[dref * 2 + 1] + oy / Hf) * Hf - 0.5f;
            const float xf = floorf(x), yf = floorf(y);
            const int x0 = (int)xf, y0 = (int)yf;
            const float fx = x - xf, fy = y - yf;
            float s = 0.f;
            const float w00 = (1.f - fx) * (1.f - fy);
            const float w10 = fx * (1.f - fy);
            const float w01 = (1.f - fx) * fy;
            const float w11 = fx * fy;
            if ((unsigned)x0 < (unsigned)Wi && (unsigned)y0 < (unsigned)Hi)
                s += w00 * (float)vb[(size_t)(base + y0 * Wi + x0) * 256];
            if ((unsigned)(x0 + 1) < (unsigned)Wi && (unsigned)y0 < (unsigned)Hi)
                s += w10 * (float)vb[(size_t)(base + y0 * Wi + x0 + 1) * 256];
            if ((unsigned)x0 < (unsigned)Wi && (unsigned)(y0 + 1) < (unsigned)Hi)
                s += w01 * (float)vb[(size_t)(base + (y0 + 1) * Wi + x0) * 256];
            if ((unsigned)(x0 + 1) < (unsigned)Wi && (unsigned)(y0 + 1) < (unsigned)Hi)
                s += w11 * (float)vb[(size_t)(base + (y0 + 1) * Wi + x0 + 1) * 256];
            acc += aw * s;
        }
    }
    out_cam[((size_t)cam * NQ + q) * 256 + h * 32 + lane] = acc;
}

// ---------------------------------------------------------------------------
// Masked mean over cameras: hit = any(bev_mask[cam,q,:4]); mean over hits.
// ---------------------------------------------------------------------------
__global__ __launch_bounds__(256) void combine_kernel(
    const float* __restrict__ out_cam, const unsigned char* __restrict__ bev_mask,
    float* __restrict__ slots)
{
    const int i = blockIdx.x * 256 + threadIdx.x;   // q*256 + e
    const int q = i >> 8;
    float sum = 0.f, cnt = 0.f;
    #pragma unroll
    for (int cam = 0; cam < CAMS; ++cam) {
        const unsigned char* mp = bev_mask + ((size_t)cam * NQ + q) * 4;
        if (mp[0] | mp[1] | mp[2] | mp[3]) {
            sum += out_cam[(size_t)cam * NQ * 256 + i];
            cnt += 1.f;
        }
    }
    slots[i] = sum / fmaxf(cnt, 1.f);
}

// ---------------------------------------------------------------------------
extern "C" void kernel_launch(void* const* d_in, const int* in_sizes, int n_in,
                              void* d_out, int out_size, void* d_ws, size_t ws_size,
                              hipStream_t stream)
{
    (void)in_sizes; (void)n_in; (void)out_size; (void)ws_size;
    const float* query      = (const float*)d_in[0];
    const float* value_feat = (const float*)d_in[2];
    const float* refpts     = (const float*)d_in[3];
    const unsigned char* bev_mask = (const unsigned char*)d_in[4];
    const float* W_value = (const float*)d_in[7];
    const float* b_value = (const float*)d_in[8];
    const float* W_off   = (const float*)d_in[9];
    const float* b_off   = (const float*)d_in[10];
    const float* W_attn  = (const float*)d_in[11];
    const float* b_attn  = (const float*)d_in[12];
    const float* W_out   = (const float*)d_in[13];
    const float* b_out   = (const float*)d_in[14];

    char* ws = (char*)d_ws;
    size_t off = 0;
    auto alloc = [&](size_t bytes) -> char* {
        off = (off + 255) & ~(size_t)255;
        char* p = ws + off;
        off += bytes;
        return p;
    };
    _Float16* Wv16    = (_Float16*)alloc((size_t)256 * 256 * 2);
    _Float16* Woa16   = (_Float16*)alloc((size_t)768 * 256 * 2);
    _Float16* Wo16    = (_Float16*)alloc((size_t)256 * 256 * 2);
    float*    bias_oa = (float*)   alloc((size_t)768 * 4);
    _Float16* val16   = (_Float16*)alloc((size_t)CAMS * LTOT * 256 * 2);
    float*    offattn = (float*)   alloc((size_t)NQ * 768 * 4);
    float*    attn    = (float*)   alloc((size_t)NQ * 256 * 4);
    float*    out_cam = (float*)   alloc((size_t)CAMS * NQ * 256 * 4);
    float*    slots   = (float*)   alloc((size_t)NQ * 256 * 4);

    // 1) weight transpose + fp16 convert + fused off/attn bias
    prep_kernel<<<1283, 256, 0, stream>>>(W_value, W_off, W_attn, b_off, b_attn,
                                          W_out, Wv16, Woa16, Wo16, bias_oa);

    // 2) value projection: per-cam [19560,256] @ [256,256] -> fp16 val
    wmma_gemm_k256<<<dim3(153, 4, CAMS), 256, 0, stream>>>(
        value_feat, Wv16, b_value, nullptr, nullptr, val16,
        LTOT, 256, (size_t)LTOT * 256, (size_t)LTOT * 256);

    // 3) fused offsets+attention projection (shared across cams): [4096,768]
    wmma_gemm_k256<<<dim3(32, 12, 1), 256, 0, stream>>>(
        query, Woa16, bias_oa, nullptr, offattn, nullptr, NQ, 768, 0, 0);

    // 4) softmax over 32 per (q, head)
    softmax_kernel<<<NQ, 256, 0, stream>>>(offattn, attn);

    // 5) deformable bilinear gather per (cam, q, head)
    sample_kernel<<<dim3(NQ, CAMS), 256, 0, stream>>>(val16, offattn, attn,
                                                      refpts, out_cam);

    // 6) masked camera mean
    combine_kernel<<<NQ, 256, 0, stream>>>(out_cam, bev_mask, slots);

    // 7) output projection + bias + residual -> d_out (fp32)
    wmma_gemm_k256<<<dim3(32, 4, 1), 256, 0, stream>>>(
        slots, Wo16, b_out, query, (float*)d_out, nullptr, NQ, 256, 0, 0);
}
